// GRAND_13975823582076
// MI455X (gfx1250) — matrix operand order repaired and checked by reading, not compiled, not run on hardware
//
#include <hip/hip_runtime.h>
#include <math.h>

typedef __attribute__((ext_vector_type(16))) __bf16 v16bf;
typedef __attribute__((ext_vector_type(8)))  float  v8f;

#define H_DIM 128

// ---------------------------------------------------------------- utilities
__global__ void zero_i32_k(int* __restrict__ p, int n) {
  int i = blockIdx.x * blockDim.x + threadIdx.x;
  if (i < n) p[i] = 0;
}

__global__ void zero_f4_k(float4* __restrict__ p, int n4) {
  int i = blockIdx.x * blockDim.x + threadIdx.x;
  if (i < n4) p[i] = make_float4(0.f, 0.f, 0.f, 0.f);
}

__global__ void copy_f4_k(const float4* __restrict__ a, float4* __restrict__ b, int n4) {
  int i = blockIdx.x * blockDim.x + threadIdx.x;
  if (i < n4) b[i] = a[i];
}

// transpose + convert fp32 W[K][N] -> bf16 Wt[N][K]
__global__ void convert_wt_k(const float* __restrict__ W, __bf16* __restrict__ Wt,
                             int K, int N) {
  int idx = blockIdx.x * blockDim.x + threadIdx.x;
  if (idx < K * N) {
    const int k = idx / N;
    const int n = idx - k * N;
    Wt[(size_t)n * K + k] = (__bf16)W[idx];
  }
}

// ---------------------------------------------------------------- degrees
__global__ void count_deg_k(const int* __restrict__ src, const int* __restrict__ dst,
                            int* __restrict__ odeg, int* __restrict__ ideg, int E) {
  int i = blockIdx.x * blockDim.x + threadIdx.x;
  if (i < E) {
    atomicAdd(&odeg[src[i]], 1);
    atomicAdd(&ideg[dst[i]], 1);
  }
}

__global__ void make_norm_k(const int* __restrict__ odeg, const int* __restrict__ ideg,
                            float* __restrict__ nsrc, float* __restrict__ ndst, int n) {
  int i = blockIdx.x * blockDim.x + threadIdx.x;
  if (i < n) {
    nsrc[i] = rsqrtf(fmaxf((float)odeg[i], 1.0f));
    ndst[i] = rsqrtf(fmaxf((float)ideg[i], 1.0f));
  }
}

// ---------------------------------------------------------------- WMMA GEMM
// D[M,N] = A[M,K] @ W[K,N] + bias[N]; A fp32, weights pre-converted to bf16
// transposed Wt[N][K]. One wave computes TN adjacent 16x16 output tiles,
// sharing the A fragment. K templated -> fully unrolled (K/32 WMMAs per tile).
// Requires M%16==0, K%32==0, N%(16*TN)==0.
template <int K, int TN>
__global__ void wmma_gemm_bias_k(const float* __restrict__ A,
                                 const __bf16* __restrict__ Wt,
                                 const float* __restrict__ bias,
                                 float* __restrict__ D,
                                 int M, int N) {
  const int lane = threadIdx.x & 31;
  const int wib  = threadIdx.x >> 5;
  const int wpb  = blockDim.x >> 5;
  const int tile = blockIdx.x * wpb + wib;
  const int tilesN = N / (16 * TN);
  const int tilesM = M >> 4;
  if (tile >= tilesM * tilesN) return;     // wave-uniform: EXEC all-ones past here
  const int tm = tile / tilesN;
  const int tn = (tile - tm * tilesN) * TN;

  const int half = lane >> 4;              // half-wave select
  const int l15  = lane & 15;
  const int rowg = tm * 16 + l15;          // A row owned by this lane
  const int akb  = half * 16;              // A: K 0..15 vs 16..31
  const int bkb  = half * 16;              // B: K 0..15 vs 16..31

  const float4* __restrict__ Ar = (const float4*)(A + (size_t)rowg * K);

  v8f acc[TN];
  v8f zero = {};
#pragma unroll
  for (int t = 0; t < TN; ++t) acc[t] = zero;

#pragma unroll
  for (int k0 = 0; k0 < K; k0 += 32) {
    // A fragment: two contiguous 8-float runs -> 4x b128 loads, pack to bf16
    const int i0 = (k0 + akb) >> 2;
    const int i1 = (k0 + akb + 8) >> 2;
    float4 p0 = Ar[i0];
    float4 p1 = Ar[i0 + 1];
    float4 p2 = Ar[i1];
    float4 p3 = Ar[i1 + 1];
    v16bf a;
    a[0]  = (__bf16)p0.x; a[1]  = (__bf16)p0.y; a[2]  = (__bf16)p0.z; a[3]  = (__bf16)p0.w;
    a[4]  = (__bf16)p1.x; a[5]  = (__bf16)p1.y; a[6]  = (__bf16)p1.z; a[7]  = (__bf16)p1.w;
    a[8]  = (__bf16)p2.x; a[9]  = (__bf16)p2.y; a[10] = (__bf16)p2.z; a[11] = (__bf16)p2.w;
    a[12] = (__bf16)p3.x; a[13] = (__bf16)p3.y; a[14] = (__bf16)p3.z; a[15] = (__bf16)p3.w;
#pragma unroll
    for (int t = 0; t < TN; ++t) {
      const int colg = (tn + t) * 16 + l15;
      // 16 consecutive bf16 K-values for this column: one 32B vector load
      v16bf b = *(const v16bf*)(Wt + (size_t)colg * K + k0 + bkb);
      acc[t] = __builtin_amdgcn_wmma_f32_16x16x32_bf16(
          /*neg_a=*/false, a, /*neg_b=*/false, b,
          /*c_mod=*/(short)0, acc[t], /*reuse_a=*/false, /*reuse_b=*/false);
    }
  }

#pragma unroll
  for (int t = 0; t < TN; ++t) {
    const int colg = (tn + t) * 16 + l15;
    const float bv = bias[colg];
#pragma unroll
    for (int r = 0; r < 8; ++r) {          // VGPR r: M = r (+8 for upper half-wave)
      const int mg = tm * 16 + r + 8 * half;
      D[(size_t)mg * N + colg] = acc[t][r] + bv;
    }
  }
}

// ---------------------------------------------------------------- propagation
// One wave per edge: gather 128-f row of src (float4/lane), scale by norm_src,
// scatter-add into dst row with L2-resident f32 atomics.
__global__ void edge_scatter_k(const float* __restrict__ h,
                               const float* __restrict__ nsrc,
                               const int* __restrict__ src,
                               const int* __restrict__ dst,
                               float* __restrict__ hnext, int E) {
  const int wave = (int)((blockIdx.x * blockDim.x + threadIdx.x) >> 5);
  const int lane = threadIdx.x & 31;
  if (wave >= E) return;                   // wave-uniform
  if (wave + 256 < E) {
    __builtin_prefetch(src + wave + 256, 0, 1);   // global_prefetch_b8
    __builtin_prefetch(dst + wave + 256, 0, 1);
  }
  const int s = src[wave];
  const int d = dst[wave];
  const float ns = nsrc[s];
  float4 v = ((const float4*)(h + (size_t)s * H_DIM))[lane];
  float* out = hnext + (size_t)d * H_DIM + lane * 4;
  unsafeAtomicAdd(out + 0, v.x * ns);
  unsafeAtomicAdd(out + 1, v.y * ns);
  unsafeAtomicAdd(out + 2, v.z * ns);
  unsafeAtomicAdd(out + 3, v.w * ns);
}

// hnext *= norm_dst[row]; hacc += hnext   (float4 over N*H)
__global__ void scale_acc_k(float4* __restrict__ hnext, const float* __restrict__ ndst,
                            float4* __restrict__ hacc, int n4) {
  int i = blockIdx.x * blockDim.x + threadIdx.x;
  if (i < n4) {
    const int row = i >> 5;                // 32 float4 per 128-f row
    const float nd = ndst[row];
    float4 v = hnext[i];
    v.x *= nd; v.y *= nd; v.z *= nd; v.w *= nd;
    hnext[i] = v;
    float4 a = hacc[i];
    a.x += v.x; a.y += v.y; a.z += v.z; a.w += v.w;
    hacc[i] = a;
  }
}

// ---------------------------------------------------------------- layer norm
// One wave per 128-f row. Optional pre-op: scale (LN0: /9) or ELU (LN1).
__global__ void fused_ln_k(const float* __restrict__ X, const float* __restrict__ g,
                           const float* __restrict__ bvec, float* __restrict__ Y,
                           int rows, float pre_scale, int do_elu) {
  const int wave = (int)((blockIdx.x * blockDim.x + threadIdx.x) >> 5);
  const int lane = threadIdx.x & 31;
  if (wave >= rows) return;                // wave-uniform
  float4 v = ((const float4*)(X + (size_t)wave * H_DIM))[lane];
  float t0, t1, t2, t3;
  if (do_elu) {
    t0 = v.x > 0.f ? v.x : expm1f(v.x);
    t1 = v.y > 0.f ? v.y : expm1f(v.y);
    t2 = v.z > 0.f ? v.z : expm1f(v.z);
    t3 = v.w > 0.f ? v.w : expm1f(v.w);
  } else {
    t0 = v.x * pre_scale; t1 = v.y * pre_scale;
    t2 = v.z * pre_scale; t3 = v.w * pre_scale;
  }
  float s  = t0 + t1 + t2 + t3;
  float s2 = t0 * t0 + t1 * t1 + t2 * t2 + t3 * t3;
#pragma unroll
  for (int off = 16; off > 0; off >>= 1) { // wave32 butterfly reduce
    s  += __shfl_xor(s,  off, 32);
    s2 += __shfl_xor(s2, off, 32);
  }
  const float m   = s * (1.0f / H_DIM);
  const float var = s2 * (1.0f / H_DIM) - m * m;   // biased var (matches jnp.var)
  const float inv = rsqrtf(var + 1e-5f);
  float4 g4 = ((const float4*)g)[lane];
  float4 b4 = ((const float4*)bvec)[lane];
  float4 o;
  o.x = (t0 - m) * inv * g4.x + b4.x;
  o.y = (t1 - m) * inv * g4.y + b4.y;
  o.z = (t2 - m) * inv * g4.z + b4.z;
  o.w = (t3 - m) * inv * g4.w + b4.w;
  ((float4*)(Y + (size_t)wave * H_DIM))[lane] = o;
}

// ---------------------------------------------------------------- row normalize
__global__ void rownorm_k(float* __restrict__ logits, int rows) {
  int r = blockIdx.x * blockDim.x + threadIdx.x;
  if (r >= rows) return;
  float4* p = (float4*)(logits + (size_t)r * 16);
  float4 a = p[0], b = p[1], c = p[2], d = p[3];
  float s = a.x*a.x + a.y*a.y + a.z*a.z + a.w*a.w
          + b.x*b.x + b.y*b.y + b.z*b.z + b.w*b.w
          + c.x*c.x + c.y*c.y + c.z*c.z + c.w*c.w
          + d.x*d.x + d.y*d.y + d.z*d.z + d.w*d.w;
  const float inv = 1.0f / fmaxf(sqrtf(s), 1e-12f);
  a.x *= inv; a.y *= inv; a.z *= inv; a.w *= inv;
  b.x *= inv; b.y *= inv; b.z *= inv; b.w *= inv;
  c.x *= inv; c.y *= inv; c.z *= inv; c.w *= inv;
  d.x *= inv; d.y *= inv; d.z *= inv; d.w *= inv;
  p[0] = a; p[1] = b; p[2] = c; p[3] = d;
}

// ---------------------------------------------------------------- launcher
extern "C" void kernel_launch(void* const* d_in, const int* in_sizes, int n_in,
                              void* d_out, int out_size, void* d_ws, size_t ws_size,
                              hipStream_t stream) {
  const float* feat0 = (const float*)d_in[0];
  const float* feat1 = (const float*)d_in[1];
  const int*   src   = (const int*)d_in[2];
  const int*   dst   = (const int*)d_in[3];
  const float* W0  = (const float*)d_in[4];
  const float* b0  = (const float*)d_in[5];
  const float* W1  = (const float*)d_in[6];
  const float* b1  = (const float*)d_in[7];
  const float* ln0g = (const float*)d_in[8];
  const float* ln0b = (const float*)d_in[9];
  const float* ln1g = (const float*)d_in[10];
  const float* ln1b = (const float*)d_in[11];
  const float* Wp  = (const float*)d_in[12];
  const float* bp  = (const float*)d_in[13];
  const float* Wo  = (const float*)d_in[14];
  const float* bo  = (const float*)d_in[15];

  const int K0 = 256, H = H_DIM, C = 16, L = 8;
  const int M0 = in_sizes[0] / K0;   // 20000
  const int M1 = in_sizes[1] / H;    // 20000
  const int Nn = M0 + M1;            // 40000
  const int E  = in_sizes[2];        // 640000

  // workspace carve-out (all offsets 16B aligned)
  char* ws = (char*)d_ws;
  int*    odeg = (int*)ws;     ws += (size_t)Nn * 4;
  int*    ideg = (int*)ws;     ws += (size_t)Nn * 4;
  float*  nsrc = (float*)ws;   ws += (size_t)Nn * 4;
  float*  ndst = (float*)ws;   ws += (size_t)Nn * 4;
  __bf16* Wt0  = (__bf16*)ws;  ws += (size_t)K0 * H * 2;   // [H][K0]
  __bf16* Wt1  = (__bf16*)ws;  ws += (size_t)H  * H * 2;   // [H][H]
  __bf16* Wtp  = (__bf16*)ws;  ws += (size_t)H  * H * 2;   // [H][H]
  __bf16* Wto  = (__bf16*)ws;  ws += (size_t)H  * C * 2;   // [C][H]
  float*  hA   = (float*)ws;   ws += (size_t)Nn * H * 4;
  float*  hB   = (float*)ws;   ws += (size_t)Nn * H * 4;
  float*  hAcc = (float*)ws;

  auto cdiv = [](int a, int b) { return (a + b - 1) / b; };
  const int n4 = Nn * H / 4;

  // degrees -> symmetric norms
  zero_i32_k<<<cdiv(2 * Nn, 256), 256, 0, stream>>>(odeg, 2 * Nn);
  count_deg_k<<<cdiv(E, 256), 256, 0, stream>>>(src, dst, odeg, ideg, E);
  make_norm_k<<<cdiv(Nn, 256), 256, 0, stream>>>(odeg, ideg, nsrc, ndst, Nn);

  // weights -> transposed bf16 (tiny, once per launch)
  convert_wt_k<<<cdiv(K0 * H, 256), 256, 0, stream>>>(W0, Wt0, K0, H);
  convert_wt_k<<<cdiv(H * H, 256), 256, 0, stream>>>(W1, Wt1, H, H);
  convert_wt_k<<<cdiv(H * H, 256), 256, 0, stream>>>(Wp, Wtp, H, H);
  convert_wt_k<<<cdiv(H * C, 256), 256, 0, stream>>>(Wo, Wto, H, C);

  // per-type input projections (WMMA), concatenated into hA
  {
    int tiles = (M0 / 16) * (H / 32);
    wmma_gemm_bias_k<256, 2><<<cdiv(tiles, 8), 256, 0, stream>>>(feat0, Wt0, b0, hA, M0, H);
  }
  {
    int tiles = (M1 / 16) * (H / 32);
    wmma_gemm_bias_k<128, 2><<<cdiv(tiles, 8), 256, 0, stream>>>(feat1, Wt1, b1,
                                                                 hA + (size_t)M0 * H, M1, H);
  }

  // h_acc = h
  copy_f4_k<<<cdiv(n4, 256), 256, 0, stream>>>((const float4*)hA, (float4*)hAcc, n4);

  // L propagation hops: h <- Ddst^-1/2 A Dsrc^-1/2 h ; h_acc += h
  float* cur = hA;
  float* nxt = hB;
  for (int hop = 0; hop < L; ++hop) {
    zero_f4_k<<<cdiv(n4, 256), 256, 0, stream>>>((float4*)nxt, n4);
    edge_scatter_k<<<cdiv(E, 8), 256, 0, stream>>>(cur, nsrc, src, dst, nxt, E);
    scale_acc_k<<<cdiv(n4, 256), 256, 0, stream>>>((float4*)nxt, ndst, (float4*)hAcc, n4);
    float* t = cur; cur = nxt; nxt = t;
  }

  // h_ = LN0(h_acc / (L+1))   -> hA
  fused_ln_k<<<cdiv(Nn, 8), 256, 0, stream>>>(hAcc, ln0g, ln0b, hA, Nn,
                                              1.0f / (float)(L + 1), 0);
  // predict layer: hB = hA @ Wp + bp  (WMMA)
  {
    int tiles = (Nn / 16) * (H / 32);
    wmma_gemm_bias_k<128, 2><<<cdiv(tiles, 8), 256, 0, stream>>>(hA, Wtp, bp, hB, Nn, H);
  }
  // hAcc = LN1(ELU(hB))
  fused_ln_k<<<cdiv(Nn, 8), 256, 0, stream>>>(hB, ln1g, ln1b, hAcc, Nn, 1.0f, 1);
  // logits = hAcc @ Wo + bo  (WMMA) -> d_out
  {
    int tiles = (Nn / 16) * (C / 16);
    wmma_gemm_bias_k<128, 1><<<cdiv(tiles, 8), 256, 0, stream>>>(hAcc, Wto, bo,
                                                                 (float*)d_out, Nn, C);
  }
  // L2 row-normalize logits in place
  rownorm_k<<<cdiv(Nn, 256), 256, 0, stream>>>((float*)d_out, Nn);
}
